// SelfAttention_64295660421124
// MI455X (gfx1250) — matrix operand order repaired and checked by reading, not compiled, not run on hardware
//
#include <hip/hip_runtime.h>

// ---------------------------------------------------------------------------
// Self-attention forward for MI455X (gfx1250, wave32, WMMA bf16 16x16x32).
//   x:[4,2048,1024] f32; W*:[1024,1024] f32; b*:[1024] f32 -> out:[4,2048,1024] f32
// Pipeline (intermediates bf16 in d_ws, ~80MB -> resident in the 192MB L2):
//   1) k_qkv    : q = xWq^T+bq, k = xWk^T+bk (bf16), v written TRANSPOSED (vT)
//                 reg-prefetch + double-buffered LDS (f32->bf16 cvt in VALU)
//   2) k_scores : P = softmax(q k^T * DIM^-0.5) (bf16); q tile staged via
//                 GLOBAL_LOAD_ASYNC_TO_LDS_B128
//   3) k_pv     : out = P @ v, async-DMA double-buffered LDS pipeline
// ---------------------------------------------------------------------------

#define DIM   1024
#define SEQ   2048
#define NB    4
#define NROW  (NB * SEQ)          // 8192 total rows of x
#define SCALE 0.03125f            // 1024^-0.5
#define LOG2E 1.44269504088896f
#define USE_ASYNC_LDS 1

typedef __bf16 bf16;
typedef __attribute__((ext_vector_type(16))) __bf16 v16bf;
typedef __attribute__((ext_vector_type(8)))  __bf16 v8bf;
typedef __attribute__((ext_vector_type(4)))  __bf16 v4bf;
typedef __attribute__((ext_vector_type(8)))  float  v8f;

static __device__ inline v16bf cat16(v8bf lo, v8bf hi) {
  return __builtin_shufflevector(lo, hi, 0,1,2,3,4,5,6,7,8,9,10,11,12,13,14,15);
}
static __device__ inline v8f vzero8() {
  v8f z;
  #pragma unroll
  for (int i = 0; i < 8; ++i) z[i] = 0.0f;
  return z;
}
static __device__ inline v8f wmma_bf16(v16bf a, v16bf b, v8f c) {
  return __builtin_amdgcn_wmma_f32_16x16x32_bf16(false, a, false, b, (short)0, c,
                                                 false, false);
}

#if USE_ASYNC_LDS
// Async DMA: 16B global -> LDS, tracked by ASYNCcnt (CDNA5 ISA).
// Generic pointers to __shared__ carry the LDS byte offset in addr[31:0]
// (flat-aperture rule, CDNA5 ISA).
static __device__ inline void async_ld_b128(void* lds, const void* gaddr) {
  const unsigned lds_addr = (unsigned)(size_t)lds;
  asm volatile("global_load_async_to_lds_b128 %0, %1, off"
               :: "v"(lds_addr), "v"(gaddr) : "memory");
}
static __device__ inline void wait_async0() {
  asm volatile("s_wait_asynccnt 0x0" ::: "memory");
}
#else
static __device__ inline void async_ld_b128(void* lds, const void* gaddr) {
  *(v8bf*)lds = *(const v8bf*)gaddr;
}
static __device__ inline void wait_async0() {}
#endif

// ---------------------------------------------------------------------------
// Kernel 1: QKV projection.  C[m,n] = sum_d x[m,d]*W[n,d] + b[n]   (NT GEMM)
// grid: (NROW/128, DIM/128, 3)   block: 256 (8 waves, 2x4 wave grid)
// Software pipeline: registers hold tile kt+32 while WMMAs consume tile kt
// from one of two LDS buffers; one barrier per iteration.
// ---------------------------------------------------------------------------
__global__ __launch_bounds__(256) void k_qkv(
    const float* __restrict__ x,
    const float* __restrict__ Wq, const float* __restrict__ bq,
    const float* __restrict__ Wk, const float* __restrict__ bk,
    const float* __restrict__ Wv, const float* __restrict__ bv,
    bf16* __restrict__ qo, bf16* __restrict__ ko, bf16* __restrict__ vTo)
{
  const int z = blockIdx.z;
  const float* __restrict__ W    = (z == 0) ? Wq : (z == 1) ? Wk : Wv;
  const float* __restrict__ bias = (z == 0) ? bq : (z == 1) ? bk : bv;

  const int m0 = blockIdx.x * 128;
  const int n0 = blockIdx.y * 128;

  __shared__ __align__(16) bf16 As[2][128][40];   // +8 pad: 16B-aligned, bank skew
  __shared__ __align__(16) bf16 Bs[2][128][40];

  const int tid  = threadIdx.x;
  const int lane = tid & 31;
  const int wave = tid >> 5;
  const int wm   = wave >> 2;           // 0..1 : 64-row slab
  const int wn   = wave & 3;            // 0..3 : 32-col slab

  v8f acc[4][2];
  #pragma unroll
  for (int i = 0; i < 4; ++i)
    #pragma unroll
    for (int j = 0; j < 2; ++j) acc[i][j] = vzero8();

  const int lr = tid >> 3;              // 0..31
  const int lc = (tid & 7) * 4;         // 0..28 step 4

  float4 xr[4], wr[4];
  auto load_regs = [&](int kt) {
    #pragma unroll
    for (int p = 0; p < 4; ++p) {
      const int r = lr + 32 * p;
      xr[p] = *(const float4*)(x + (size_t)(m0 + r) * DIM + kt + lc);
      wr[p] = *(const float4*)(W + (size_t)(n0 + r) * DIM + kt + lc);
    }
  };
  auto store_lds = [&](int buf) {
    #pragma unroll
    for (int p = 0; p < 4; ++p) {
      const int r = lr + 32 * p;
      v4bf va; va[0]=(bf16)xr[p].x; va[1]=(bf16)xr[p].y; va[2]=(bf16)xr[p].z; va[3]=(bf16)xr[p].w;
      *(v4bf*)&As[buf][r][lc] = va;
      v4bf vb; vb[0]=(bf16)wr[p].x; vb[1]=(bf16)wr[p].y; vb[2]=(bf16)wr[p].z; vb[3]=(bf16)wr[p].w;
      *(v4bf*)&Bs[buf][r][lc] = vb;
    }
  };

  load_regs(0);
  store_lds(0);
  int buf = 0;
  for (int kt = 0; kt < DIM; kt += 32) {
    __syncthreads();
    const bool more = (kt + 32 < DIM);
    if (more) load_regs(kt + 32);       // global loads overlap the WMMAs below

    const int koff = (lane >> 4) * 8;
    v16bf bfrag[2];
    #pragma unroll
    for (int j = 0; j < 2; ++j) {
      const int bn = wn * 32 + j * 16 + (lane & 15);
      const int bk_ = (lane >> 4) * 8;
      bfrag[j] = cat16(*(const v8bf*)&Bs[buf][bn][bk_],
                       *(const v8bf*)&Bs[buf][bn][bk_ + 16]);
    }
    #pragma unroll
    for (int i = 0; i < 4; ++i) {
      const int ar = wm * 64 + i * 16 + (lane & 15);
      const v16bf afrag = cat16(*(const v8bf*)&As[buf][ar][koff],
                                *(const v8bf*)&As[buf][ar][16 + koff]);
      #pragma unroll
      for (int j = 0; j < 2; ++j) acc[i][j] = wmma_bf16(afrag, bfrag[j], acc[i][j]);
    }
    if (more) store_lds(buf ^ 1);       // other buffer: no second barrier needed
    buf ^= 1;
  }

  // ---- bias + store (C/D layout: elem r of lane l -> M=r+8*(l>=16), N=l&15) ----
  #pragma unroll
  for (int i = 0; i < 4; ++i) {
    #pragma unroll
    for (int j = 0; j < 2; ++j) {
      const int ncol = n0 + wn * 32 + j * 16 + (lane & 15);
      const float ba = bias[ncol];
      const int mbase = m0 + wm * 64 + i * 16 + ((lane >> 4) * 8);
      #pragma unroll
      for (int r = 0; r < 8; ++r) {
        const float val = acc[i][j][r] + ba;
        const int m = mbase + r;
        if (z == 2) {
          const int b = m >> 11, s = m & 2047;
          vTo[((size_t)b * DIM + ncol) * SEQ + s] = (bf16)val;
        } else {
          bf16* dst = (z == 0) ? qo : ko;
          dst[(size_t)m * DIM + ncol] = (bf16)val;
        }
      }
    }
  }
}

// ---------------------------------------------------------------------------
// Kernel 2: scores + softmax.  One block = 16 q-rows x all 2048 keys.
// 8 waves, wave w owns keys [256w, 256w+256): 16 v8f accumulators.
// grid: (SEQ/16, NB)   block: 256
// ---------------------------------------------------------------------------
__global__ __launch_bounds__(256) void k_scores(
    const bf16* __restrict__ qb, const bf16* __restrict__ kb, bf16* __restrict__ P)
{
  const int b  = blockIdx.y;
  const int q0 = blockIdx.x * 16;
  const bf16* __restrict__ qrow = qb + ((size_t)b * SEQ + q0) * DIM;
  const bf16* __restrict__ kmat = kb + (size_t)b * SEQ * DIM;
  bf16* __restrict__ Pb = P + (size_t)b * SEQ * SEQ;

  __shared__ __align__(16) bf16 qs[16][DIM + 8];   // 16 x 1032 bf16 = 33 KB
  __shared__ float red[8][16];

  const int tid  = threadIdx.x;
  const int lane = tid & 31;
  const int wave = tid >> 5;
  const int key0 = wave * 256;

  // stage q tile via async DMA: 2048 16B chunks / 256 threads = 8 each
  for (int c = tid; c < 16 * 128; c += 256) {
    const int r = c >> 7, cc = (c & 127) * 8;
    async_ld_b128(&qs[r][cc], &qrow[(size_t)r * DIM + cc]);
  }
  wait_async0();
  __syncthreads();

  v8f acc[16];
  #pragma unroll
  for (int t = 0; t < 16; ++t) acc[t] = vzero8();

  const int koff = (lane >> 4) * 8;
  const int arow = lane & 15;
  for (int dc = 0; dc < DIM; dc += 32) {
    const v16bf afrag = cat16(*(const v8bf*)&qs[arow][dc + koff],
                              *(const v8bf*)&qs[arow][dc + 16 + koff]);
    #pragma unroll
    for (int t = 0; t < 16; ++t) {
      const int key = key0 + t * 16 + (lane & 15);
      const bf16* kp = kmat + (size_t)key * DIM + dc + (lane >> 4) * 8;
      const v16bf bfrag = cat16(*(const v8bf*)kp, *(const v8bf*)(kp + 16));
      acc[t] = wmma_bf16(afrag, bfrag, acc[t]);
    }
  }

  // ---- row max over this wave's 256 keys ----
  float rmax[8];
  #pragma unroll
  for (int r = 0; r < 8; ++r) {
    float m = acc[0][r];
    #pragma unroll
    for (int t = 1; t < 16; ++t) m = fmaxf(m, acc[t][r]);
    #pragma unroll
    for (int off = 1; off <= 8; off <<= 1) m = fmaxf(m, __shfl_xor(m, off, 32));
    rmax[r] = m;
  }
  if ((lane & 15) == 0) {
    const int half = lane >> 4;
    #pragma unroll
    for (int r = 0; r < 8; ++r) red[wave][half * 8 + r] = rmax[r];
  }
  __syncthreads();
  float gmax[8];
  #pragma unroll
  for (int r = 0; r < 8; ++r) {
    const int row = (lane >> 4) * 8 + r;
    float m = red[0][row];
    #pragma unroll
    for (int w2 = 1; w2 < 8; ++w2) m = fmaxf(m, red[w2][row]);
    gmax[r] = m;
  }
  __syncthreads();

  // ---- exp + row sum (softmax of SCALE*s == exp2(SCALE*LOG2E*(s-max))) ----
  const float cc = SCALE * LOG2E;
  float rsum[8];
  #pragma unroll
  for (int r = 0; r < 8; ++r) rsum[r] = 0.0f;
  #pragma unroll
  for (int t = 0; t < 16; ++t)
    #pragma unroll
    for (int r = 0; r < 8; ++r) {
      const float p = exp2f((acc[t][r] - gmax[r]) * cc);
      acc[t][r] = p;
      rsum[r] += p;
    }
  #pragma unroll
  for (int r = 0; r < 8; ++r)
    #pragma unroll
    for (int off = 1; off <= 8; off <<= 1) rsum[r] += __shfl_xor(rsum[r], off, 32);
  if ((lane & 15) == 0) {
    const int half = lane >> 4;
    #pragma unroll
    for (int r = 0; r < 8; ++r) red[wave][half * 8 + r] = rsum[r];
  }
  __syncthreads();
  float rinv[8];
  #pragma unroll
  for (int r = 0; r < 8; ++r) {
    const int row = (lane >> 4) * 8 + r;
    float s = red[0][row];
    #pragma unroll
    for (int w2 = 1; w2 < 8; ++w2) s += red[w2][row];
    rinv[r] = 1.0f / s;
  }

  // ---- normalize + store P (bf16) ----
  #pragma unroll
  for (int t = 0; t < 16; ++t) {
    const int key = key0 + t * 16 + (lane & 15);
    #pragma unroll
    for (int r = 0; r < 8; ++r) {
      const int row = (lane >> 4) * 8 + r;
      Pb[(size_t)(q0 + row) * SEQ + key] = (bf16)(acc[t][r] * rinv[r]);
    }
  }
}

// ---------------------------------------------------------------------------
// Kernel 3: out = P @ V via NT GEMM (B = vT, K = keys contiguous).
// Async-DMA double-buffered LDS: issue tile i+1 right after the barrier, then
// compute tile i -> DMA overlaps the entire WMMA phase; 1 barrier/iteration.
// grid: (SEQ/128, DIM/128, NB)   block: 256 (8 waves, 2x4)
// ---------------------------------------------------------------------------
__global__ __launch_bounds__(256) void k_pv(
    const bf16* __restrict__ P, const bf16* __restrict__ vT, float* __restrict__ out)
{
  const int b  = blockIdx.z;
  const int m0 = blockIdx.x * 128;     // q rows within batch
  const int n0 = blockIdx.y * 128;     // d cols
  const bf16* __restrict__ Pb  = P  + (size_t)b * SEQ * SEQ;
  const bf16* __restrict__ vTb = vT + (size_t)b * DIM * SEQ;
  float* __restrict__ ob = out + (size_t)b * SEQ * DIM;

  __shared__ __align__(16) bf16 As[2][128][40];
  __shared__ __align__(16) bf16 Bs[2][128][40];

  const int tid  = threadIdx.x;
  const int lane = tid & 31;
  const int wave = tid >> 5;
  const int wm   = wave >> 2;
  const int wn   = wave & 3;

  v8f acc[4][2];
  #pragma unroll
  for (int i = 0; i < 4; ++i)
    #pragma unroll
    for (int j = 0; j < 2; ++j) acc[i][j] = vzero8();

  // per-tile staging: 128 rows x 4 chunks(16B) per matrix = 512 DMAs / 256 thr
  auto issue_tile = [&](int kt, int bufi) {
    #pragma unroll
    for (int it = 0; it < 2; ++it) {
      const int c = tid + it * 256;
      const int r = c >> 2, cc = (c & 3) * 8;
      async_ld_b128(&As[bufi][r][cc], &Pb [(size_t)(m0 + r) * SEQ + kt + cc]);
      async_ld_b128(&Bs[bufi][r][cc], &vTb[(size_t)(n0 + r) * SEQ + kt + cc]);
    }
  };

  issue_tile(0, 0);
  int buf = 0;
  for (int kt = 0; kt < SEQ; kt += 32) {
    wait_async0();                     // tile kt landed in LDS (this wave's DMAs)
    __syncthreads();                   // all waves' DMAs visible; old reads done
    if (kt + 32 < SEQ) issue_tile(kt + 32, buf ^ 1);

    const int koff = (lane >> 4) * 8;
    v16bf bfrag[2];
    #pragma unroll
    for (int j = 0; j < 2; ++j) {
      const int bn = wn * 32 + j * 16 + (lane & 15);
      const int bk_ = (lane >> 4) * 8;
      bfrag[j] = cat16(*(const v8bf*)&Bs[buf][bn][bk_],
                       *(const v8bf*)&Bs[buf][bn][bk_ + 16]);
    }
    #pragma unroll
    for (int i = 0; i < 4; ++i) {
      const int ar = wm * 64 + i * 16 + (lane & 15);
      const v16bf afrag = cat16(*(const v8bf*)&As[buf][ar][koff],
                                *(const v8bf*)&As[buf][ar][16 + koff]);
      #pragma unroll
      for (int j = 0; j < 2; ++j) acc[i][j] = wmma_bf16(afrag, bfrag[j], acc[i][j]);
    }
    buf ^= 1;
  }

  #pragma unroll
  for (int i = 0; i < 4; ++i)
    #pragma unroll
    for (int j = 0; j < 2; ++j) {
      const int ncol = n0 + wn * 32 + j * 16 + (lane & 15);
      const int mbase = m0 + wm * 64 + i * 16 + ((lane >> 4) * 8);
      #pragma unroll
      for (int r = 0; r < 8; ++r)
        ob[(size_t)(mbase + r) * DIM + ncol] = acc[i][j][r];
    }
}

// ---------------------------------------------------------------------------
extern "C" void kernel_launch(void* const* d_in, const int* in_sizes, int n_in,
                              void* d_out, int out_size, void* d_ws, size_t ws_size,
                              hipStream_t stream)
{
  (void)in_sizes; (void)n_in; (void)out_size; (void)ws_size;
  const float* x  = (const float*)d_in[0];
  const float* Wq = (const float*)d_in[1];
  const float* bq = (const float*)d_in[2];
  const float* Wk = (const float*)d_in[3];
  const float* bk = (const float*)d_in[4];
  const float* Wv = (const float*)d_in[5];
  const float* bv = (const float*)d_in[6];

  // workspace (bf16): q[8192x1024] k[8192x1024] vT[4x1024x2048] P[4x2048x2048]
  bf16* qb = (bf16*)d_ws;
  bf16* kb = qb + (size_t)NROW * DIM;
  bf16* vT = kb + (size_t)NROW * DIM;
  bf16* Pm = vT + (size_t)NROW * DIM;

  k_qkv   <<<dim3(NROW / 128, DIM / 128, 3), 256, 0, stream>>>(
      x, Wq, bq, Wk, bk, Wv, bv, qb, kb, vT);
  k_scores<<<dim3(SEQ / 16, NB), 256, 0, stream>>>(qb, kb, Pm);
  k_pv    <<<dim3(SEQ / 128, DIM / 128, NB), 256, 0, stream>>>(
      Pm, vT, (float*)d_out);
}
